// SocialLSTM_68272800137829
// MI455X (gfx1250) — matrix operand order repaired and checked by reading, not compiled, not run on hardware
//
#include <hip/hip_runtime.h>
#include <hip/hip_bf16.h>

// ---------------------------------------------------------------------------
// Social LSTM on gfx1250 (MI455X).  T=64 sequential steps, 3 kernels/step.
// GEMMs use v_wmma_f32_16x16x32_f16 (wave32).  grid values are 0/1 -> f16
// exact; h/c state fp32; weights converted once to K-major f16 (L2-resident).
// Dominant HBM traffic = 256 MB grid stream (non-temporal, double-buffered).
// B-matrix (masked h, 64 KB) is staged into LDS per workgroup with CDNA5
// async global->LDS copies (ASYNCcnt), so the WMMA inner loop is LDS-fed.
// ---------------------------------------------------------------------------

#define T_  64
#define A_  256
#define GG_ 16
#define H_  128
#define PE_ 64
#define GE_ 64
#define IN_ 128
#define G4_ 512            // 4*H
#define HMP_ 136           // padded LDS row pitch (halfs) for hm tile

typedef _Float16 h16 __attribute__((ext_vector_type(16)));
typedef _Float16 h8  __attribute__((ext_vector_type(8)));
typedef float    f8  __attribute__((ext_vector_type(8)));
typedef float    f4v __attribute__((ext_vector_type(4)));

__device__ __forceinline__ h16 cat8(h8 lo, h8 hi) {
  return __builtin_shufflevector(lo, hi, 0,1,2,3,4,5,6,7,8,9,10,11,12,13,14,15);
}

// A-matrix fragment (16x32 f16, row-major source, row stride ld halfs).
// ISA layout: lane l<16 -> M=l, K={k0..k0+7, k0+16..k0+23}; lanes 16-31 -> +8.
__device__ __forceinline__ h16 load_afrag_f16(const _Float16* base, int row0,
                                              int ld, int k0, int lane) {
  int m = lane & 15, sel = lane >> 4;
  const _Float16* p = base + (size_t)(row0 + m) * ld + k0 + sel * 8;
  h8 lo = *(const h8*)p;
  h8 hi = *(const h8*)(p + 16);
  return cat8(lo, hi);
}

// Same fragment from fp32 source (grid tensor): non-temporal stream + cvt.
__device__ __forceinline__ h16 load_afrag_f32_nt(const float* base, int row0,
                                                 int ld, int k0, int lane) {
  int m = lane & 15, sel = lane >> 4;
  const float* p = base + (size_t)(row0 + m) * ld + k0 + sel * 8;
  f4v a0 = __builtin_nontemporal_load((const f4v*)p);
  f4v a1 = __builtin_nontemporal_load((const f4v*)(p + 4));
  f4v b0 = __builtin_nontemporal_load((const f4v*)(p + 16));
  f4v b1 = __builtin_nontemporal_load((const f4v*)(p + 20));
  h16 r;
#pragma unroll
  for (int i = 0; i < 4; ++i) {
    r[i]      = (_Float16)a0[i];
    r[4 + i]  = (_Float16)a1[i];
    r[8 + i]  = (_Float16)b0[i];
    r[12 + i] = (_Float16)b1[i];
  }
  return r;
}

// B-matrix fragment (32x16 f16) from K-major global source [K][ldb]:
// lane l holds N=n0+l%16, K={k0..k0+7, k0+16..k0+23} (+8 for lanes 16-31).
__device__ __forceinline__ h16 load_bfrag(const _Float16* base, int k0,
                                          int ldb, int n0, int lane) {
  const _Float16* p = base + (size_t)(k0 + (lane >> 4) * 8) * ldb + n0 + (lane & 15);
  h16 r;
#pragma unroll
  for (int s = 0; s < 8; ++s) {
    r[s]     = p[(size_t)s * ldb];
    r[8 + s] = p[(size_t)(16 + s) * ldb];
  }
  return r;
}

__device__ __forceinline__ f8 wmma_f16(h16 a, h16 b, f8 c) {
  return __builtin_amdgcn_wmma_f32_16x16x32_f16(false, a, false, b,
                                                (short)0, c, false, false);
}

// CDNA5 async global->LDS copy (16 B per lane), tracked by ASYNCcnt.
__device__ __forceinline__ void async_g2l_b128(unsigned lds_byte_off,
                                               const void* gptr) {
  asm volatile("global_load_async_to_lds_b128 %0, %1, off"
               :: "v"(lds_byte_off), "v"((unsigned long long)gptr)
               : "memory");
}
__device__ __forceinline__ void wait_async0() {
  asm volatile("s_wait_asynccnt 0x0" ::: "memory");
}

__device__ __forceinline__ float sigmoidf_(float x) {
  return 1.0f / (1.0f + __expf(-x));
}
__device__ __forceinline__ float tanhf_(float x) {
  float e = __expf(-2.0f * fabsf(x));
  float r = (1.0f - e) / (1.0f + e);
  return x < 0.0f ? -r : r;
}

// ---------------------------------------------------------------------------
// One-time prep: f16 K-major weight transposes + seed h/c state (in d_out).
// ---------------------------------------------------------------------------
__global__ __launch_bounds__(256) void k_prep(
    const float* __restrict__ soc_W, const float* __restrict__ W_ih,
    const float* __restrict__ W_hh, const float* __restrict__ h0,
    const float* __restrict__ c0, _Float16* __restrict__ socWT,
    _Float16* __restrict__ WihT, _Float16* __restrict__ WhhT,
    float* __restrict__ stateH, float* __restrict__ stateC) {
  int i = blockIdx.x * 256 + threadIdx.x;        // 131072 threads
  {                                              // socWT[k][64], k<2048
    int k = i >> 6, n = i & 63;
    socWT[i] = (_Float16)soc_W[(size_t)n * (GG_ * H_) + k];
  }
  if (i < H_ * G4_) {                            // WihT/WhhT [128][512]
    int k = i >> 9, n = i & 511;
    WihT[i] = (_Float16)W_ih[(size_t)n * H_ + k];
    WhhT[i] = (_Float16)W_hh[(size_t)n * H_ + k];
  }
  if (i < A_ * H_) {
    stateH[i] = h0[i];
    stateC[i] = c0[i];
  }
}

// ---------------------------------------------------------------------------
// Per-step prologue: pos MLP (relu) -> inp[:, :64]; f16 snapshots of h, m*h.
// ---------------------------------------------------------------------------
__global__ __launch_bounds__(256) void k_step_pre(
    const float* __restrict__ x_t, const unsigned char* __restrict__ m_t,
    const float* __restrict__ pos_W, const float* __restrict__ pos_b,
    const float* __restrict__ stateH, _Float16* __restrict__ inp16,
    _Float16* __restrict__ h16buf, _Float16* __restrict__ hm16) {
  int a = threadIdx.x;
  float x0 = x_t[a * 2], x1 = x_t[a * 2 + 1];
  float mf = m_t[a] ? 1.0f : 0.0f;
#pragma unroll 4
  for (int j = 0; j < PE_; ++j) {
    float s = x0 * pos_W[j * 2] + x1 * pos_W[j * 2 + 1] + pos_b[j];
    inp16[a * IN_ + j] = (_Float16)(s > 0.0f ? s : 0.0f);
  }
#pragma unroll 4
  for (int j = 0; j < H_; ++j) {
    float v = stateH[a * H_ + j];
    h16buf[a * H_ + j] = (_Float16)v;
    hm16[a * H_ + j]   = (_Float16)(v * mf);
  }
}

// ---------------------------------------------------------------------------
// Heavy kernel: social = grid(4096x256) @ hm(256x128) (per 16-agent block:
// 256x128 tile -> LDS), then soc = relu(social_r(16x2048) @ socWT + b) fused.
// hm is async-staged into LDS (padded pitch -> conflict-free 16-bit B-fragment loads);
// the grid A-stream is double-buffered non-temporal VMEM.
// ---------------------------------------------------------------------------
__global__ __launch_bounds__(256) void k_social(
    const float* __restrict__ gridt, const _Float16* __restrict__ hm16,
    const _Float16* __restrict__ socWT, const float* __restrict__ soc_b,
    _Float16* __restrict__ inp16) {
  __shared__ __attribute__((aligned(16))) _Float16 soc_lds[256 * 128]; // 64 KB
  __shared__ __attribute__((aligned(16))) _Float16 hm_lds[256 * HMP_]; // 68 KB
  int lane = threadIdx.x & 31, w = threadIdx.x >> 5;
  int a0 = blockIdx.x * 16;
  int r0blk = a0 * GG_;                          // first grid row of block

  // ---- stage hm (256 x 128 f16) into LDS with async global->LDS copies ----
  {
    int r = threadIdx.x;                         // one row per thread
    unsigned lbase = (unsigned)(size_t)&hm_lds[r * HMP_];
    const char* gbase = (const char*)(hm16 + r * H_);
#pragma unroll
    for (int j = 0; j < 16; ++j)                 // 16 x 16 B = 256 B row
      async_g2l_b128(lbase + j * 16, gbase + j * 16);
    wait_async0();
  }
  __syncthreads();

  // ---- social phase: each wave owns 2 M-tiles (16 rows each) ----
  // Live set: acc[8] (64 VGPR) + a_cur/a_nxt (16) + 1 LDS b-frag (8).
#pragma unroll 1
  for (int i = 0; i < 2; ++i) {
    int mt = w * 2 + i;
    int r0 = r0blk + mt * 16;
    f8 acc[8] = {};
    h16 a_cur = load_afrag_f32_nt(gridt, r0, A_, 0, lane);
#pragma unroll 1
    for (int kc = 0; kc < 8; ++kc) {
      h16 a_nxt = a_cur;
      if (kc < 7)                                // prefetch next HBM chunk
        a_nxt = load_afrag_f32_nt(gridt, r0, A_, (kc + 1) * 32, lane);
#pragma unroll
      for (int nt = 0; nt < 8; ++nt) {
        const _Float16* bp = &hm_lds[(unsigned)(kc * 32 + (lane >> 4) * 8) * HMP_ + nt * 16 + (lane & 15)];
        h16 bfrag;
#pragma unroll
        for (int s = 0; s < 8; ++s) {
          bfrag[s]     = bp[s * HMP_];
          bfrag[8 + s] = bp[(16 + s) * HMP_];
        }
        acc[nt] = wmma_f16(a_cur, bfrag, acc[nt]);
      }
      a_cur = a_nxt;
    }
    int sel = lane >> 4, col = lane & 15;
#pragma unroll
    for (int nt = 0; nt < 8; ++nt)
#pragma unroll
      for (int v = 0; v < 8; ++v) {
        int rl = mt * 16 + v + 8 * sel;          // row within 256-row tile
        soc_lds[rl * H_ + nt * 16 + col] = (_Float16)acc[nt][v];
      }
  }
  __syncthreads();

  // ---- soc phase: 4 waves, one 16-col N-tile each, K=2048 from LDS ----
  if (w < 4) {
    f8 acc = {};
    int n0 = w * 16;
    int m = lane & 15, sel = lane >> 4;
#pragma unroll 4
    for (int kc = 0; kc < 64; ++kc) {
      int k0 = kc * 32;
      const _Float16* ap = &soc_lds[m * (GG_ * H_) + k0 + sel * 8];
      h16 afrag = cat8(*(const h8*)ap, *(const h8*)(ap + 16));
      h16 bfrag = load_bfrag(socWT, k0, GE_, n0, lane);
      acc = wmma_f16(afrag, bfrag, acc);
    }
    int nn = n0 + (lane & 15);
    float bias = soc_b[nn];
#pragma unroll
    for (int v = 0; v < 8; ++v) {
      int al = v + 8 * sel;
      float s = acc[v] + bias;
      s = s > 0.0f ? s : 0.0f;
      inp16[(a0 + al) * IN_ + PE_ + nn] = (_Float16)s;
    }
  }
}

// ---------------------------------------------------------------------------
// Gates (WMMA) + LSTM elementwise + masked state update + output projection.
// Wave w owns H-columns [w*16, w*16+16) across the 4 gate blocks i,f,g,o.
// ---------------------------------------------------------------------------
__global__ __launch_bounds__(256) void k_gates(
    const _Float16* __restrict__ inp16, const _Float16* __restrict__ h16buf,
    const _Float16* __restrict__ WihT, const _Float16* __restrict__ WhhT,
    const float* __restrict__ b_ih, const float* __restrict__ b_hh,
    const unsigned char* __restrict__ m_t, const float* __restrict__ out_W,
    const float* __restrict__ out_b, float* __restrict__ stateH,
    float* __restrict__ stateC, float* __restrict__ hnew,
    float* __restrict__ out_t) {
  int lane = threadIdx.x & 31, w = threadIdx.x >> 5;
  int a0 = blockIdx.x * 16;
  int nw = w * 16;

  f8 acc[4] = {};
#pragma unroll 1
  for (int kc = 0; kc < 4; ++kc) {
    int k0 = kc * 32;
    h16 ai = load_afrag_f16(inp16, a0, IN_, k0, lane);
    h16 ah = load_afrag_f16(h16buf, a0, H_, k0, lane);
#pragma unroll
    for (int gi = 0; gi < 4; ++gi) {
      int gcol = gi * H_ + nw;
      acc[gi] = wmma_f16(ai, load_bfrag(WihT, k0, G4_, gcol, lane), acc[gi]);
      acc[gi] = wmma_f16(ah, load_bfrag(WhhT, k0, G4_, gcol, lane), acc[gi]);
    }
  }

  int sel = lane >> 4, col = lane & 15;
  float bias[4];
#pragma unroll
  for (int gi = 0; gi < 4; ++gi)
    bias[gi] = b_ih[gi * H_ + nw + col] + b_hh[gi * H_ + nw + col];

#pragma unroll
  for (int v = 0; v < 8; ++v) {
    int a = a0 + v + 8 * sel;
    int hc = nw + col;
    float iv = sigmoidf_(acc[0][v] + bias[0]);
    float fv = sigmoidf_(acc[1][v] + bias[1]);
    float gv = tanhf_(acc[2][v] + bias[2]);
    float ov = sigmoidf_(acc[3][v] + bias[3]);
    float c_old = stateC[a * H_ + hc];
    float c_new = fv * c_old + iv * gv;
    float h_new = ov * tanhf_(c_new);
    hnew[a * H_ + hc] = h_new;                   // pred uses unmasked h_new
    if (m_t[a]) {                                // masked state carry
      stateH[a * H_ + hc] = h_new;
      stateC[a * H_ + hc] = c_new;
    }
  }
  __syncthreads();

  if (threadIdx.x < 32) {                        // 16 agents x 2 outputs
    int al = threadIdx.x >> 1, oc = threadIdx.x & 1;
    int a = a0 + al;
    float s = out_b[oc];
#pragma unroll 4
    for (int hh = 0; hh < H_; ++hh) s += hnew[a * H_ + hh] * out_W[oc * H_ + hh];
    out_t[a * 2 + oc] = m_t[a] ? s : 0.0f;
  }
}

// ---------------------------------------------------------------------------
extern "C" void kernel_launch(void* const* d_in, const int* in_sizes, int n_in,
                              void* d_out, int out_size, void* d_ws,
                              size_t ws_size, hipStream_t stream) {
  const float* input_tensor  = (const float*)d_in[0];          // (T,A,2)
  const float* grid_tensor   = (const float*)d_in[1];          // (T,A,GG,A)
  const float* hidden0       = (const float*)d_in[2];          // (A,H)
  const float* cell0         = (const float*)d_in[3];          // (A,H)
  const unsigned char* mask  = (const unsigned char*)d_in[4];  // (T,A) bool
  const float* pos_W = (const float*)d_in[5];
  const float* pos_b = (const float*)d_in[6];
  const float* soc_W = (const float*)d_in[7];
  const float* soc_b = (const float*)d_in[8];
  const float* W_ih  = (const float*)d_in[9];
  const float* W_hh  = (const float*)d_in[10];
  const float* b_ih  = (const float*)d_in[11];
  const float* b_hh  = (const float*)d_in[12];
  const float* out_W = (const float*)d_in[13];
  const float* out_b = (const float*)d_in[14];

  float* outs   = (float*)d_out;                 // (T,A,2)
  float* stateH = outs + T_ * A_ * 2;            // (A,H) final h lives here
  float* stateC = stateH + A_ * H_;              // (A,H) final c lives here

  char* ws = (char*)d_ws;
  _Float16* socWT  = (_Float16*)ws;                      // 2048*64   (256 KB)
  _Float16* WihT   = (_Float16*)(ws + 262144);           // 128*512   (128 KB)
  _Float16* WhhT   = (_Float16*)(ws + 262144 + 131072);
  _Float16* h16buf = (_Float16*)(ws + 262144 + 2 * 131072);           // 64 KB
  _Float16* hm16   = (_Float16*)(ws + 262144 + 2 * 131072 + 65536);
  _Float16* inp16  = (_Float16*)(ws + 262144 + 2 * 131072 + 2 * 65536);
  float*    hnew   = (float*)(ws + 262144 + 2 * 131072 + 3 * 65536);  // 128 KB

  k_prep<<<512, 256, 0, stream>>>(soc_W, W_ih, W_hh, hidden0, cell0, socWT,
                                  WihT, WhhT, stateH, stateC);

  for (int t = 0; t < T_; ++t) {
    k_step_pre<<<1, 256, 0, stream>>>(input_tensor + (size_t)t * A_ * 2,
                                      mask + (size_t)t * A_, pos_W, pos_b,
                                      stateH, inp16, h16buf, hm16);
    k_social<<<16, 256, 0, stream>>>(grid_tensor + (size_t)t * A_ * GG_ * A_,
                                     hm16, socWT, soc_b, inp16);
    k_gates<<<16, 256, 0, stream>>>(inp16, h16buf, WihT, WhhT, b_ih, b_hh,
                                    mask + (size_t)t * A_, out_W, out_b,
                                    stateH, stateC, hnew, outs + (size_t)t * A_ * 2);
  }
}
